// SparseGO_71700184039905
// MI455X (gfx1250) — matrix operand order repaired and checked by reading, not compiled, not run on hardware
//
#include <hip/hip_runtime.h>
#include <hip/hip_bf16.h>

typedef __attribute__((ext_vector_type(16))) __bf16 v16bf;
typedef __attribute__((ext_vector_type(8)))  __bf16 v8bf;
typedef __attribute__((ext_vector_type(2)))  __bf16 v2bf;
typedef __attribute__((ext_vector_type(8)))  float  v8f;

#define BM 128
#define BN 128
#define BK 32
#define LDSS 40              // ushort stride per row (80 bytes, 16B aligned)
#define BUFELTS (BM * LDSS)  // one buffer (A or B), in ushorts
#define BATCH_DIM 4096
#define K_DIM 20000
#define N_DIM 5000

// two fp32 -> packed bf16 (RNE) in one instruction where available
__device__ __forceinline__ unsigned int cvt_pk_bf16(float a, float b) {
#if __has_builtin(__builtin_amdgcn_cvt_pk_bf16_f32)
  v2bf p = __builtin_amdgcn_cvt_pk_bf16_f32(a, b);
  return __builtin_bit_cast(unsigned int, p);
#else
  v2bf p; p[0] = (__bf16)a; p[1] = (__bf16)b;
  return __builtin_bit_cast(unsigned int, p);
#endif
}

__global__ __launch_bounds__(256) void sparsego_wmma_bf16(
    const float* __restrict__ X,   // [4096, 20000] row-major
    const float* __restrict__ W,   // [20000, 5000] row-major
    const float* __restrict__ Msk, // [20000, 5000] row-major
    float* __restrict__ C)         // [4096, 5000]  row-major
{
  // double-buffered: 2 x (10KB A + 10KB B) bf16 tiles (padded) = 40 KB of 320 KB
  __shared__ unsigned short As[2 * BUFELTS];
  __shared__ unsigned short Bs[2 * BUFELTS];

  const int tid   = threadIdx.x;
  const int lane  = tid & 31;
  const int wave  = tid >> 5;
  const int waveM = wave & 1;   // 0..1 -> 64-row slab
  const int waveN = wave >> 1;  // 0..3 -> 32-col slab

  const int blockM = blockIdx.y * BM;
  const int blockN = blockIdx.x * BN;

  // global->reg staging maps
  const int xRow0 = tid >> 3;          // 0..31, +32 each rep (4 reps -> 128 rows)
  const int xK    = (tid & 7) * 4;     // 8 float4 chunks cover K=32
  const int wK0   = (tid >> 5) * 4;    // wave id *4: reps give k = wK0 + i (consecutive)
  const int wN    = (tid & 31) * 4;    // 32 float4 chunks cover N=128
  const int gN    = blockN + wN;
  const bool nOK  = (gN < N_DIM);

  v8f acc[4][2];
#pragma unroll
  for (int mi = 0; mi < 4; ++mi)
#pragma unroll
    for (int ni = 0; ni < 2; ++ni)
      acc[mi][ni] = (v8f){0.f,0.f,0.f,0.f,0.f,0.f,0.f,0.f};

  float4 xreg[4];
  float  wreg[4][4];                   // [k-rep][n-component]

  const int ksteps = K_DIM / BK;       // 625 exact

  auto load_tile = [&](int ks) {
    const long kBase = (long)ks * BK;
#pragma unroll
    for (int i = 0; i < 4; ++i) {
      const long row = blockM + xRow0 + 32 * i;
      xreg[i] = *(const float4*)&X[row * K_DIM + kBase + xK];
    }
#pragma unroll
    for (int i = 0; i < 4; ++i) {
      if (nOK) {
        const long gk = kBase + wK0 + i;       // 4 consecutive k rows per thread
        const float4 w4 = *(const float4*)&W[gk * N_DIM + gN];
        const float4 m4 = *(const float4*)&Msk[gk * N_DIM + gN];
        wreg[i][0] = w4.x * m4.x; wreg[i][1] = w4.y * m4.y;
        wreg[i][2] = w4.z * m4.z; wreg[i][3] = w4.w * m4.w;
      } else {
        wreg[i][0] = wreg[i][1] = wreg[i][2] = wreg[i][3] = 0.f;
      }
    }
    // hint the next tile into L2 (speculative)
    if (ks + 1 < ksteps) {
      const long kNext = (long)(ks + 1) * BK;
      __builtin_prefetch(&X[(long)(blockM + xRow0) * K_DIM + kNext + xK], 0, 1);
      if (nOK) __builtin_prefetch(&W[(kNext + wK0) * N_DIM + gN], 0, 1);
    }
  };

  load_tile(0);

  const int khalf = lane >> 4;     // 0|1
  const int l16   = lane & 15;

  for (int ks = 0; ks < ksteps; ++ks) {
    const int buf = (ks & 1) * BUFELTS;
    unsigned short* Ab = &As[buf];
    unsigned short* Bb = &Bs[buf];

    // ---- stage regs -> LDS[buf] (fp32 -> bf16, packed b64 stores) ----
#pragma unroll
    for (int i = 0; i < 4; ++i) {
      const int row = xRow0 + 32 * i;
      uint2 p;
      p.x = cvt_pk_bf16(xreg[i].x, xreg[i].y);
      p.y = cvt_pk_bf16(xreg[i].z, xreg[i].w);
      *(uint2*)&Ab[row * LDSS + xK] = p;
    }
#pragma unroll
    for (int j = 0; j < 4; ++j) {                  // transpose: column wN+j gets 4 consecutive k
      uint2 p;
      p.x = cvt_pk_bf16(wreg[0][j], wreg[1][j]);
      p.y = cvt_pk_bf16(wreg[2][j], wreg[3][j]);
      *(uint2*)&Bb[(wN + j) * LDSS + wK0] = p;
    }
    __syncthreads();   // single rendezvous per K-step (double-buffered LDS)

    if (ks + 1 < ksteps) load_tile(ks + 1);        // overlap next global loads with WMMA

    // ---- build fragments per ISA 16-bit layouts ----
    v16bf afrag[4], bfrag[2];
#pragma unroll
    for (int mi = 0; mi < 4; ++mi) {
      const int base = (waveM * 64 + mi * 16 + l16) * LDSS + khalf * 8;
      v8bf lo = *(const v8bf*)&Ab[base];           // K = khalf*8 .. +7      (VGPR0-3)
      v8bf hi = *(const v8bf*)&Ab[base + 16];      // K = 16+khalf*8 .. +7   (VGPR4-7)
      afrag[mi] = __builtin_shufflevector(lo, hi, 0,1,2,3,4,5,6,7,8,9,10,11,12,13,14,15);
    }
#pragma unroll
    for (int ni = 0; ni < 2; ++ni) {
      const int base = (waveN * 32 + ni * 16 + l16) * LDSS + khalf * 8;
      v8bf lo = *(const v8bf*)&Bb[base];           // K = khalf*8 .. +7
      v8bf hi = *(const v8bf*)&Bb[base + 16];      // K = 16+khalf*8 .. +7
      bfrag[ni] = __builtin_shufflevector(lo, hi, 0,1,2,3,4,5,6,7,8,9,10,11,12,13,14,15);
    }

#pragma unroll
    for (int mi = 0; mi < 4; ++mi)
#pragma unroll
      for (int ni = 0; ni < 2; ++ni)
        acc[mi][ni] = __builtin_amdgcn_wmma_f32_16x16x32_bf16(
            false, afrag[mi], false, bfrag[ni],
            (short)0, acc[mi][ni], false, false);
    // no trailing barrier: next iteration writes the other buffer; the
    // s_wait_dscnt before WMMA consumption orders this buffer's reads
    // ahead of the next barrier, so iteration ks+2 cannot overwrite early.
  }

  // ---- store C (16x16 f32 C/D layout: VGPR r -> M = r + 8*(lane>=16), N = lane&15) ----
#pragma unroll
  for (int mi = 0; mi < 4; ++mi) {
#pragma unroll
    for (int ni = 0; ni < 2; ++ni) {
      const int row0 = blockM + waveM * 64 + mi * 16 + (khalf ? 8 : 0);
      const int col  = blockN + waveN * 32 + ni * 16 + l16;
      if (col < N_DIM) {
#pragma unroll
        for (int r = 0; r < 8; ++r)
          C[(long)(row0 + r) * N_DIM + col] = acc[mi][ni][r];
      }
    }
  }
}

extern "C" void kernel_launch(void* const* d_in, const int* in_sizes, int n_in,
                              void* d_out, int out_size, void* d_ws, size_t ws_size,
                              hipStream_t stream) {
  const float* X   = (const float*)d_in[0];   // [4096, 20000]
  const float* W   = (const float*)d_in[1];   // [20000, 5000]
  const float* Msk = (const float*)d_in[2];   // [20000, 5000]
  float* C = (float*)d_out;                   // [4096, 5000]

  dim3 grid((N_DIM + BN - 1) / BN, BATCH_DIM / BM);  // (40, 32)
  dim3 block(256);
  sparsego_wmma_bf16<<<grid, block, 0, stream>>>(X, W, Msk, C);
}
